// SCDecoder_22686017257976
// MI455X (gfx1250) — matrix operand order repaired and checked by reading, not compiled, not run on hardware
//
#include <hip/hip_runtime.h>

// CDNA5 / gfx1250 neural SC polar decoder.
//  kernel 1: pack_kernel   - f32 weights -> f16 WMMA-B fragment-major tiles in ws, + c0/c1 tables
//  kernel 2: obs_kernel    - e = mlp(y)  (WMMA), e stored f16 in ws
//  kernel 3: decode_kernel - one workgroup per codeword; whole SC recursion in-LDS with WMMA
// NOTE: decode/obs use dynamic LDS (>64KB is legal on CDNA5 WGP: 320KB); compile-only target.

typedef _Float16 f16;
typedef __attribute__((ext_vector_type(16))) _Float16 v16h;
typedef __attribute__((ext_vector_type(8)))  _Float16 v8h;
typedef __attribute__((ext_vector_type(8)))  float    v8f;

#define B_  256
#define N_  256
#define D_  128
#define H_  256

// row offset of level-L activation buffer in the LDS stack, L in [1,8]
__device__ __forceinline__ int lvloff(int L) { return 256 - (512 >> L); }

// ---------------------------------------------------------------------------
// Tiled GEMM:  Out(rows x N) = act( A(rows x K) @ Bpack + bias [+ c_{usel[row]}] )
// A: f16 (LDS or global), row-major, stride lda halves. K multiple of 32.
// Bpack: fragment-major f16 tiles: tile (kt,ct) at (kt*N/16+ct)*512, lane L holds
//        halves h=0..15 with K = kt*32 + (L>>4)*8 + (h&7) + (h>>3)*16, col = ct*16 + (L&15).
// Out: f16, row-major, stride ldo. 256-thread block = 8 waves; wave-uniform tile loop
// so EXEC is all-ones at every v_wmma (ISA requirement).
// ---------------------------------------------------------------------------
__device__ __forceinline__ void gemm_wmma(
    const f16* __restrict__ A, int lda, int rows, int K,
    const f16* __restrict__ Bp, int N,
    const float* __restrict__ bias,
    const float* __restrict__ c0, const float* __restrict__ c1,
    const int* __restrict__ usel,
    f16* __restrict__ Out, int ldo, bool relu, bool guard)
{
  const int lane = threadIdx.x & 31;
  const int wv   = threadIdx.x >> 5;     // 0..7
  const int m    = lane & 15;
  const int g    = lane >> 4;
  const int Nt   = N >> 4;
  const int Mt   = (rows + 15) >> 4;
  const int Kt   = K >> 5;
  const int total = Mt * Nt;
  for (int t = wv; t < total; t += 8) {
    const int mt = t / Nt;
    const int nt = t - mt * Nt;
    v8f acc = {};
    const f16* pa = A + (size_t)(mt * 16 + m) * lda + g * 8;
    const f16* pb = Bp + ((size_t)nt << 9) + (lane << 4);
    for (int kt = 0; kt < Kt; ++kt) {
      // A 16x32 f16 fragment: lane m holds K = g*8 + [0,8) and 16 + g*8 + [0,8)
      v8h alo = *(const v8h*)(pa);
      v8h ahi = *(const v8h*)(pa + 16);
      v16h av = __builtin_shufflevector(alo, ahi, 0,1,2,3,4,5,6,7,8,9,10,11,12,13,14,15);
      v16h bv = *(const v16h*)(pb);      // one contiguous 32B load per lane
      acc = __builtin_amdgcn_wmma_f32_16x16x32_f16(
          false, av, false, bv, (short)0, acc, false, false);
      pa += 32;
      pb += (size_t)Nt << 9;
    }
    const int col = nt * 16 + m;
    const float bb = bias[col];
    float a0 = 0.f, a1 = 0.f;
    if (usel) { a0 = c0[col]; a1 = c1[col]; }
#pragma unroll
    for (int j = 0; j < 8; ++j) {
      const int row = mt * 16 + g * 8 + j;   // C/D layout: VGPR j -> M = g*8+j
      float v = acc[j] + bb;
      if (usel) v += (usel[row] ? a1 : a0);
      if (relu) v = v > 0.f ? v : 0.f;
      if (!guard || row < rows) Out[(size_t)row * ldo + col] = (f16)v;
    }
  }
}

// ---------------------------------------------------------------------------
// Weight packer + c-tables.  One thread per packed element.
// ---------------------------------------------------------------------------
__device__ __forceinline__ void pack_one(f16* dst, const float* src,
                                         int e, int N, int Ksrc)
{
  const int tile   = e >> 9;
  const int within = e & 511;
  const int lane   = within >> 4;
  const int h      = within & 15;
  const int Nt     = N >> 4;
  const int kt     = tile / Nt;
  const int ct     = tile - kt * Nt;
  const int k      = kt * 32 + (lane >> 4) * 8 + (h & 7) + (h >> 3) * 16;
  const int c      = ct * 16 + (lane & 15);
  const float v    = (k < Ksrc) ? src[(size_t)k * N + c] : 0.f;
  dst[e] = (f16)v;
}

__global__ __launch_bounds__(256) void pack_kernel(
    const float* __restrict__ Wobs1, const float* __restrict__ Wobs2,
    const float* __restrict__ Wchk1, const float* __restrict__ Wchk2,
    const float* __restrict__ Wbit1, const float* __restrict__ Wbit2,
    const float* __restrict__ Wlabel,
    f16* obs1p, f16* obs2p, f16* chk1p, f16* chk2p,
    f16* bit1p, f16* bit2p, float* c0, float* c1)
{
  const int gtid = blockIdx.x * 256 + threadIdx.x;
  if      (gtid <   8192) pack_one(obs1p, Wobs1, gtid,          256,   4); // K pad 4->32
  else if (gtid <  40960) pack_one(obs2p, Wobs2, gtid -   8192, 128, 256);
  else if (gtid < 106496) pack_one(chk1p, Wchk1, gtid -  40960, 256, 256);
  else if (gtid < 139264) pack_one(chk2p, Wchk2, gtid - 106496, 128, 256);
  else if (gtid < 204800) pack_one(bit1p, Wbit1, gtid - 139264, 256, 256); // rows 0..255 of W_bit1
  else if (gtid < 237568) pack_one(bit2p, Wbit2, gtid - 204800, 128, 256);
  else if (gtid < 238080) {
    // c_s[j] = sum_d W_label[s][d] * W_bit1[256+d][j]   (u_emb contribution)
    const int i = gtid - 237568;
    const int s = i >> 8, j = i & 255;
    float acc = 0.f;
    for (int d = 0; d < 128; ++d)
      acc += Wlabel[s * 128 + d] * Wbit1[(size_t)(256 + d) * 256 + j];
    (s ? c1 : c0)[j] = acc;
  }
}

// ---------------------------------------------------------------------------
// obs MLP: e = relu(y @ W1 + b1) @ W2 + b2, 128 rows per block, y padded K 4->32
// ---------------------------------------------------------------------------
__global__ __launch_bounds__(256) void obs_kernel(
    const float* __restrict__ y,
    const f16* __restrict__ obs1p, const float* __restrict__ b1,
    const f16* __restrict__ obs2p, const float* __restrict__ b2,
    f16* __restrict__ e_ws)
{
  extern __shared__ char smem[];
  f16* yl = (f16*)smem;                 // 128 x 32
  f16* h  = (f16*)(smem + 128 * 32 * 2); // 128 x 256
  const int r0 = blockIdx.x * 128;
  for (int i = threadIdx.x; i < 128 * 32; i += 256) yl[i] = (f16)0.f;
  __syncthreads();
  for (int i = threadIdx.x; i < 128 * 4; i += 256) {
    const int r = i >> 2, c = i & 3;
    yl[r * 32 + c] = (f16)y[(size_t)(r0 + r) * 4 + c];
  }
  __syncthreads();
  gemm_wmma(yl, 32, 128, 32, obs1p, 256, b1, nullptr, nullptr, nullptr,
            h, 256, true, false);
  __syncthreads();
  gemm_wmma(h, 256, 128, 256, obs2p, 128, b2, nullptr, nullptr, nullptr,
            e_ws + (size_t)r0 * D_, 128, false, false);
}

// ---------------------------------------------------------------------------
// SC decode: one workgroup (256 threads / 8 waves) per codeword.
// LDS: act stack levels 1..8 (255x128 f16) | hstage 128x256 f16 | x[255] | u1[255]
// ---------------------------------------------------------------------------
__global__ __launch_bounds__(256) void decode_kernel(
    const f16* __restrict__ e_ws,
    const f16* __restrict__ chk1p, const f16* __restrict__ chk2p,
    const f16* __restrict__ bit1p, const f16* __restrict__ bit2p,
    const float* __restrict__ b_chk1, const float* __restrict__ b_chk2,
    const float* __restrict__ b_bit1, const float* __restrict__ b_bit2,
    const float* __restrict__ c0, const float* __restrict__ c1,
    const float* __restrict__ Wl1, const float* __restrict__ bl1,
    const float* __restrict__ Wl2, const float* __restrict__ bl2,
    const int* __restrict__ f, float* __restrict__ out)
{
  extern __shared__ char smem[];
  f16* act    = (f16*)smem;             // 32640 halves (levels 1..8)
  f16* hstage = (f16*)(smem + 65280);   // 32768 halves (hidden staging / llr scratch)
  int* xArr   = (int*)(smem + 130816);  // x[L] at lvloff(L), L=1..8
  int* u1s    = (int*)(smem + 131840);  // u1[L] (left-child x) at lvloff(L+1), L=0..7
  const int b   = blockIdx.x;
  const int tid = threadIdx.x;
  const f16* eroot = e_ws + (size_t)b * N_ * D_;

  for (int leaf = 0; leaf < N_; ++leaf) {
    int Lstart;
    if (leaf == 0) {
      Lstart = 0;                        // chk all the way down
    } else {
      const int z  = __ffs(leaf) - 1;    // trailing zeros
      const int Ls = 7 - z;              // node switching to its right child: bit MLP
      {
        const int rows = 256 >> (Ls + 1);
        const f16* A = (Ls == 0) ? eroot : (act + (size_t)lvloff(Ls) * D_);
        gemm_wmma(A, 256, rows, 256, bit1p, 256, b_bit1,
                  c0, c1, u1s + lvloff(Ls + 1), hstage, 256, true, false);
        __syncthreads();
        gemm_wmma(hstage, 256, rows, 256, bit2p, 128, b_bit2,
                  nullptr, nullptr, nullptr,
                  act + (size_t)lvloff(Ls + 1) * D_, 128, false, true);
        __syncthreads();
      }
      Lstart = Ls + 1;
    }
    for (int L = Lstart; L < 8; ++L) {   // chk MLPs down the fresh left path
      const int rows = 256 >> (L + 1);
      const f16* A = (L == 0) ? eroot : (act + (size_t)lvloff(L) * D_);
      gemm_wmma(A, 256, rows, 256, chk1p, 256, b_chk1,
                nullptr, nullptr, nullptr, hstage, 256, true, false);
      __syncthreads();
      gemm_wmma(hstage, 256, rows, 256, chk2p, 128, b_chk2,
                nullptr, nullptr, nullptr,
                act + (size_t)lvloff(L + 1) * D_, 128, false, true);
      __syncthreads();
    }

    // ---- leaf: llr MLP (1x128 -> 256 -> 2), softmax, hard decision ----
    {
      const f16* e8 = act + (size_t)lvloff(8) * D_;
      float* r0 = (float*)hstage;        // reuse staging as f32 scratch
      float* r1 = r0 + 256;
      float acc = bl1[tid];
      for (int d = 0; d < 128; ++d) acc += (float)e8[d] * Wl1[d * 256 + tid];
      acc = acc > 0.f ? acc : 0.f;
      r0[tid] = acc * Wl2[tid * 2 + 0];
      r1[tid] = acc * Wl2[tid * 2 + 1];
      __syncthreads();
      for (int s = 128; s > 0; s >>= 1) {
        if (tid < s) { r0[tid] += r0[tid + s]; r1[tid] += r1[tid + s]; }
        __syncthreads();
      }
      if (tid == 0) {
        const float l0 = r0[0] + bl2[0];
        const float l1 = r1[0] + bl2[1];
        const float mx = l0 > l1 ? l0 : l1;
        const float e0 = __expf(l0 - mx), e1 = __expf(l1 - mx);
        const float inv = 1.f / (e0 + e1);
        const int hd = (l1 > l0) ? 1 : 0;          // argmax (ties -> 0)
        const int fr = f[b * N_ + leaf];
        const int u  = (fr == 2) ? hd : fr;
        out[b * N_ + leaf]                      = (float)u;   // uhat
        out[B_ * N_ + (b * N_ + leaf) * 2 + 0]  = e0 * inv;   // p
        out[B_ * N_ + (b * N_ + leaf) * 2 + 1]  = e1 * inv;
        xArr[lvloff(8)] = u;
      }
      __syncthreads();
    }

    // ---- combine partial sums up for every trailing-one level ----
    int tc = 0;
    for (int j = 0; j < 8; ++j) {
      if (((leaf >> j) & 1) == 0) break;
      const int P = 7 - j;               // parent level whose right child completed
      if (P >= 1) {
        const int cn = 256 >> (P + 1);
        if (tid < cn) {
          const int u1v = u1s[lvloff(P + 1) + tid];
          const int u2v = xArr[lvloff(P + 1) + tid];
          xArr[lvloff(P) + 2 * tid]     = (u1v + u2v) & 1;  // v_xor
          xArr[lvloff(P) + 2 * tid + 1] = u2v;
        }
      }
      __syncthreads();
      ++tc;
    }
    const int Lf = 8 - tc;               // deepest now-complete node: a left child
    if (leaf != 255 && Lf >= 1) {
      const int sz = 256 >> Lf;
      if (tid < sz) u1s[lvloff(Lf) + tid] = xArr[lvloff(Lf) + tid];
    }
    __syncthreads();
  }
}

// ---------------------------------------------------------------------------
extern "C" void kernel_launch(void* const* d_in, const int* in_sizes, int n_in,
                              void* d_out, int out_size, void* d_ws, size_t ws_size,
                              hipStream_t stream)
{
  (void)in_sizes; (void)n_in; (void)out_size; (void)ws_size;
  const float* y      = (const float*)d_in[0];
  const int*   f      = (const int*)  d_in[1];
  const float* Wobs1  = (const float*)d_in[2];
  const float* bobs1  = (const float*)d_in[3];
  const float* Wobs2  = (const float*)d_in[4];
  const float* bobs2  = (const float*)d_in[5];
  const float* Wllr1  = (const float*)d_in[6];
  const float* bllr1  = (const float*)d_in[7];
  const float* Wllr2  = (const float*)d_in[8];
  const float* bllr2  = (const float*)d_in[9];
  const float* Wchk1  = (const float*)d_in[10];
  const float* bchk1  = (const float*)d_in[11];
  const float* Wchk2  = (const float*)d_in[12];
  const float* bchk2  = (const float*)d_in[13];
  const float* Wbit1  = (const float*)d_in[14];
  const float* bbit1  = (const float*)d_in[15];
  const float* Wbit2  = (const float*)d_in[16];
  const float* bbit2  = (const float*)d_in[17];
  const float* Wlabel = (const float*)d_in[18];

  // workspace layout (~17.3 MB)
  char* ws   = (char*)d_ws;
  f16* e_ws  = (f16*)ws;                          // 65536 x 128 f16 = 16 MiB
  f16* obs1p = (f16*)(ws + 16777216);             //  8192 halves
  f16* obs2p = obs1p + 8192;                      // 32768
  f16* chk1p = obs2p + 32768;                     // 65536
  f16* chk2p = chk1p + 65536;                     // 32768
  f16* bit1p = chk2p + 32768;                     // 65536
  f16* bit2p = bit1p + 65536;                     // 32768
  float* c0  = (float*)(bit2p + 32768);           // 256 f32
  float* c1  = c0 + 256;                          // 256 f32

  pack_kernel<<<930, 256, 0, stream>>>(Wobs1, Wobs2, Wchk1, Wchk2, Wbit1, Wbit2,
                                       Wlabel, obs1p, obs2p, chk1p, chk2p,
                                       bit1p, bit2p, c0, c1);
  obs_kernel<<<512, 256, 73728, stream>>>(y, obs1p, bobs1, obs2p, bobs2, e_ws);
  decode_kernel<<<256, 256, 132864, stream>>>(e_ws, chk1p, chk2p, bit1p, bit2p,
                                              bchk1, bchk2, bbit1, bbit2, c0, c1,
                                              Wllr1, bllr1, Wllr2, bllr2,
                                              f, (float*)d_out);
}
